// Repro_72026601554106
// MI455X (gfx1250) — 8-bit matrix operand order repaired and checked by reading, compiled, not run on hardware
//
#include <hip/hip_runtime.h>
#include <stdint.h>

typedef int v8i __attribute__((ext_vector_type(8)));
typedef int v4i __attribute__((ext_vector_type(4)));

#define M_DIM 8192
#define K_DIM 4096
#define N_DIM 12288

#define BM 128
#define BN 256
#define BK 64
#define LDA 80
#define LDB 80
#define NK  (K_DIM / BK)

static_assert(M_DIM % BM == 0, "");
static_assert(N_DIM % BN == 0, "");
static_assert(K_DIM % BK == 0, "");

static __device__ __forceinline__ void wait_asynccnt0() {
#if __has_builtin(__builtin_amdgcn_s_wait_asynccnt)
    __builtin_amdgcn_s_wait_asynccnt(0);
#else
    asm volatile("s_wait_asynccnt 0x0" ::: "memory");
#endif
}

__global__ __launch_bounds__(256, 1)
void int8_qkv_gemm_dequant(const int8_t* __restrict__ W,
                           const float*  __restrict__ clampv,
                           const int8_t* __restrict__ X,
                           float*        __restrict__ Y)
{
    __shared__ __attribute__((aligned(16))) int8_t As[2][BM * LDA];
    __shared__ __attribute__((aligned(16))) int8_t Bs[2][BN * LDB];

    const int tid  = threadIdx.x;
    const int lane = tid & 31;
    const int wave = tid >> 5;
    const int wm   = wave >> 2;
    const int wn   = wave & 3;

    const int block_m = blockIdx.y * BM;
    const int block_n = blockIdx.x * BN;

    auto async_load_a = [&](int kt, int buf) {
#pragma unroll
        for (int u = 0; u < 2; ++u) {
            const int c   = tid + u * 256;
            const int row = c >> 2;
            const int ko  = (c & 3) << 4;
            const int8_t* gp = X + (size_t)(block_m + row) * K_DIM
                                 + (size_t)kt * BK + ko;
            const uint32_t lds_off =
                (uint32_t)(uintptr_t)(&As[buf][row * LDA + ko]);
            asm volatile("global_load_async_to_lds_b128 %0, %1, off"
                         :: "v"(lds_off), "v"(gp) : "memory");
        }
    };

    uint32_t b_reg[4][4];

    auto load_global_b = [&](int kt) {
#pragma unroll
        for (int u = 0; u < 4; ++u) {
            const int t  = tid + u * 256;
            const int kc = t >> 6;
            const int nc = t & 63;
            const int k0 = kc << 2;
            const int n0 = nc << 2;
            const uint8_t* base = (const uint8_t*)W
                + (size_t)((size_t)kt * BK + k0) * N_DIM + block_n + n0;
            uint32_t r0 = *(const uint32_t*)(base + 0 * (size_t)N_DIM);
            uint32_t r1 = *(const uint32_t*)(base + 1 * (size_t)N_DIM);
            uint32_t r2 = *(const uint32_t*)(base + 2 * (size_t)N_DIM);
            uint32_t r3 = *(const uint32_t*)(base + 3 * (size_t)N_DIM);
            uint32_t p01_lo = __builtin_amdgcn_perm(r1, r0, 0x05010400u);
            uint32_t p01_hi = __builtin_amdgcn_perm(r1, r0, 0x07030602u);
            uint32_t p23_lo = __builtin_amdgcn_perm(r3, r2, 0x05010400u);
            uint32_t p23_hi = __builtin_amdgcn_perm(r3, r2, 0x07030602u);
            b_reg[u][0] = __builtin_amdgcn_perm(p23_lo, p01_lo, 0x05040100u);
            b_reg[u][1] = __builtin_amdgcn_perm(p23_lo, p01_lo, 0x07060302u);
            b_reg[u][2] = __builtin_amdgcn_perm(p23_hi, p01_hi, 0x05040100u);
            b_reg[u][3] = __builtin_amdgcn_perm(p23_hi, p01_hi, 0x07060302u);
        }
    };

    auto store_lds_b = [&](int buf) {
#pragma unroll
        for (int u = 0; u < 4; ++u) {
            const int t  = tid + u * 256;
            const int kc = t >> 6;
            const int nc = t & 63;
            const int k0 = kc << 2;
            const int n0 = nc << 2;
#pragma unroll
            for (int j = 0; j < 4; ++j)
                *(uint32_t*)(&Bs[buf][(n0 + j) * LDB + k0]) = b_reg[u][j];
        }
    };

    v8i acc[4][4] = {};

    async_load_a(0, 0);
    load_global_b(0);
    store_lds_b(0);
    wait_asynccnt0();
    __syncthreads();

    for (int kt = 0; kt < NK; ++kt) {
        const int  cur      = kt & 1;
        const int  nxt      = cur ^ 1;
        const bool has_next = (kt + 1) < NK;

        if (has_next) {
            async_load_a(kt + 1, nxt);
            load_global_b(kt + 1);
        }

        v8i a_frag[4];
#pragma unroll
        for (int i = 0; i < 4; ++i) {
            const int row = wm * 64 + i * 16 + (lane & 15);
            const int kh  = (lane >> 4) << 4;
            const int8_t* p = &As[cur][row * LDA + kh];
            uint64_t* d = (uint64_t*)&a_frag[i];
            d[0] = *(const uint64_t*)(p + 0);
            d[1] = *(const uint64_t*)(p + 8);
            d[2] = *(const uint64_t*)(p + 32);
            d[3] = *(const uint64_t*)(p + 40);
        }
        v8i b_frag[4];
#pragma unroll
        for (int j = 0; j < 4; ++j) {
            const int col = wn * 64 + j * 16 + (lane & 15);
            const int kh  = (lane >> 4) << 4;
            const int8_t* p = &Bs[cur][col * LDB + kh];
            v4i* d = (v4i*)&b_frag[j];
            d[0] = *(const v4i*)(p + 0);
            d[1] = *(const v4i*)(p + 32);
        }

#pragma unroll
        for (int i = 0; i < 4; ++i)
#pragma unroll
            for (int j = 0; j < 4; ++j)
                acc[i][j] = __builtin_amdgcn_wmma_i32_16x16x64_iu8(
                     true, a_frag[i],
                     true, b_frag[j],
                    acc[i][j],
                     false,  false);

        __syncthreads();
        if (has_next) {
            store_lds_b(nxt);
            wait_asynccnt0();
            __syncthreads();
        }
    }

    const float inv127 = 1.0f / 127.0f;
#pragma unroll
    for (int i = 0; i < 4; ++i) {
        const int rbase = block_m + wm * 64 + i * 16 + ((lane >> 4) << 3);
#pragma unroll
        for (int r = 0; r < 8; ++r) {
            const int   row = rbase + r;
            const float s   = clampv[row] * inv127;
#pragma unroll
            for (int j = 0; j < 4; ++j) {
                const int col = block_n + wn * 64 + j * 16 + (lane & 15);
                Y[(size_t)row * N_DIM + col] = (float)acc[i][j][r] * s;
            }
        }
    }
}

extern "C" void kernel_launch(void* const* d_in, const int* in_sizes, int n_in,
                              void* d_out, int out_size, void* d_ws, size_t ws_size,
                              hipStream_t stream) {
    (void)in_sizes; (void)n_in; (void)out_size; (void)d_ws; (void)ws_size;
    const int8_t* W      = (const int8_t*)d_in[0];
    const float*  clampv = (const float*)d_in[1];
    const int8_t* X      = (const int8_t*)d_in[2];
    float*        Y      = (float*)d_out;

    dim3 grid(N_DIM / BN, M_DIM / BM);
    int8_qkv_gemm_dequant<<<grid, dim3(256), 0, stream>>>(W, clampv, X, Y);
}
